// my_lstm_regressor_6347961664097
// MI455X (gfx1250) — matrix operand order repaired and checked by reading, not compiled, not run on hardware
//
#include <hip/hip_runtime.h>
#include <math.h>

// Problem dims (match reference)
#define T_STEPS   2048
#define NBATCH    256
#define ISZ       256
#define HSZ       512
#define GSZ       2048          // 4*HSZ gate columns
#define KTOT      768           // ISZ + HSZ fused K
#define KC        32            // bf16 WMMA K
#define NWG       32            // workgroups in persistent grid
#define NH        16            // hidden units owned per WG (HSZ/NWG)
#define NTHREADS  256           // 8 wave32

typedef __attribute__((ext_vector_type(16))) __bf16 v16bf;
typedef __attribute__((ext_vector_type(8)))  __bf16 v8bf;
typedef __attribute__((ext_vector_type(2)))  __bf16 v2bf;
typedef __attribute__((ext_vector_type(8)))  float  v8f;

#define A_STRIDE_B 80                       // staged A row: 32 bf16 + 16B pad
#define A_BUF_B    (NBATCH * A_STRIDE_B)    // 20480 bytes per ping-pong buffer

__device__ __forceinline__ unsigned f2bf(float x) {
  unsigned u = __float_as_uint(x);
  return (u + 0x7FFFu + ((u >> 16) & 1u)) >> 16;   // RNE truncate to bf16
}

__device__ __forceinline__ unsigned pack_bf16(float a, float b) {
#if __has_builtin(__builtin_amdgcn_cvt_pk_bf16_f32)
  union { v2bf v; unsigned u; } cv;
  cv.v = __builtin_amdgcn_cvt_pk_bf16_f32(a, b);
  return cv.u;
#else
  return f2bf(a) | (f2bf(b) << 16);
#endif
}

__device__ __forceinline__ float fast_rcp(float x) {
#if __has_builtin(__builtin_amdgcn_rcpf)
  return __builtin_amdgcn_rcpf(x);
#else
  return 1.0f / x;
#endif
}
__device__ __forceinline__ float fast_exp2(float x) {
#if __has_builtin(__builtin_amdgcn_exp2f)
  return __builtin_amdgcn_exp2f(x);
#else
  return exp2f(x);
#endif
}
// Branchless activations (no EXEC divergence, 1-ulp v_exp/v_rcp)
__device__ __forceinline__ float sigm(float x) {
  return fast_rcp(1.0f + fast_exp2(-1.4426950408889634f * x));
}
__device__ __forceinline__ float tanh_fast(float x) {
  float xc = fminf(fmaxf(x, -15.0f), 15.0f);      // keep exp2 arg finite
  return 1.0f - 2.0f * fast_rcp(1.0f + fast_exp2(2.8853900817779268f * xc));
}

__device__ __forceinline__ v16bf join16(v8bf lo, v8bf hi) {
  v16bf r;
#pragma unroll
  for (int i = 0; i < 8; ++i) { r[i] = lo[i]; r[i + 8] = hi[i]; }
  return r;
}

// Async global->LDS copy (CDNA5): tracked with ASYNCcnt.
#define ASYNC_COPY_B128(ldsaddr, gptr)                                         \
  asm volatile("global_load_async_to_lds_b128 %0, %1, off"                     \
               :: "v"((unsigned)(ldsaddr)),                                    \
                  "v"((unsigned long long)(uintptr_t)(gptr))                   \
               : "memory")

__device__ __forceinline__ void async_wait0() {
#if __has_builtin(__builtin_amdgcn_s_wait_asynccnt)
  __builtin_amdgcn_s_wait_asynccnt(0);
#else
  asm volatile("s_wait_asynccnt 0x0" ::: "memory");
#endif
}

// Device-wide sense barrier across the 32 persistent workgroups.
__device__ __forceinline__ void grid_barrier(unsigned* bar, unsigned expect) {
  __threadfence();          // release: push h stores to L2
  __syncthreads();
  if (threadIdx.x == 0) {
    unsigned a = __hip_atomic_fetch_add(&bar[0], 1u, __ATOMIC_ACQ_REL,
                                        __HIP_MEMORY_SCOPE_AGENT);
    if (a == NWG - 1u) {
      __hip_atomic_store(&bar[0], 0u, __ATOMIC_RELAXED, __HIP_MEMORY_SCOPE_AGENT);
      __hip_atomic_store(&bar[1], expect + 1u, __ATOMIC_RELEASE,
                         __HIP_MEMORY_SCOPE_AGENT);
    } else {
      while (__hip_atomic_load(&bar[1], __ATOMIC_ACQUIRE,
                               __HIP_MEMORY_SCOPE_AGENT) <= expect)
        __builtin_amdgcn_s_sleep(8);
    }
  }
  __syncthreads();
  __threadfence();          // acquire: invalidate L0 so fresh h is loaded
}

// ---------------------------------------------------------------------------
// Phase 0: [W_ih | W_hh] -> bf16 [GSZ x KTOT], combined bias, barrier reset.
// ---------------------------------------------------------------------------
__global__ void lstm_prep(const float* __restrict__ W_ih,
                          const float* __restrict__ W_hh,
                          const float* __restrict__ b_ih,
                          const float* __restrict__ b_hh,
                          unsigned short* __restrict__ Wb,
                          float* __restrict__ biasc,
                          unsigned* __restrict__ bar)
{
  unsigned gid = blockIdx.x * blockDim.x + threadIdx.x;
  if (gid == 0) { bar[0] = 0u; bar[1] = 0u; }
  if (gid < GSZ) biasc[gid] = b_ih[gid] + b_hh[gid];
  size_t stride = (size_t)gridDim.x * blockDim.x;
  for (size_t i = gid; i < (size_t)GSZ * KTOT; i += stride) {
    size_t r = i / KTOT;
    size_t k = i - r * KTOT;
    float v = (k < ISZ) ? W_ih[r * ISZ + k] : W_hh[r * HSZ + (k - ISZ)];
    Wb[i] = (unsigned short)f2bf(v);
  }
}

// ---------------------------------------------------------------------------
// Phase 1: persistent LSTM rollout. WG w owns hidden units [16w, 16w+16).
// Per step: [256 x 768] @ [768 x 64] bf16 WMMA GEMM + fused cell update.
// h ping-pong buffers live in global memory as bf16.
// ---------------------------------------------------------------------------
__global__ void __launch_bounds__(NTHREADS)
lstm_persistent(const float* __restrict__ inputs,
                const float* __restrict__ h0,
                const float* __restrict__ c0,
                const unsigned short* __restrict__ Wb,
                const float* __restrict__ biasc,
                unsigned short* __restrict__ hbuf0,
                unsigned short* __restrict__ hbuf1,
                unsigned* __restrict__ bar)
{
  __shared__ char  smA[2][A_BUF_B];            // ping-pong bf16 A chunks
  __shared__ float smC[NBATCH * NH];           // cell state slice [256 x 16]

  const int tid  = threadIdx.x;
  const int lane = tid & 31;
  const int wid  = tid >> 5;
  const int jj   = lane & 15;
  const int hid0 = blockIdx.x * NH;
  const int rb   = wid * 32;                   // wave's 32 batch rows

  // low 32 bits of a generic LDS pointer == LDS byte offset (aperture rule)
  const unsigned smAbase[2] = { (unsigned)(uintptr_t)&smA[0][0],
                                (unsigned)(uintptr_t)&smA[1][0] };

  // init cell-state slice and bf16 h buffer slice from c0/h0
  for (int idx = tid; idx < NBATCH * NH; idx += NTHREADS) {
    int b = idx >> 4, j = idx & 15;
    smC[idx] = c0[(size_t)b * HSZ + hid0 + j];
    hbuf0[(size_t)b * HSZ + hid0 + j] =
        (unsigned short)f2bf(h0[(size_t)b * HSZ + hid0 + j]);
  }

  const unsigned short* wrow[4];
#pragma unroll
  for (int nt = 0; nt < 4; ++nt)
    wrow[nt] = Wb + (size_t)(nt * HSZ + hid0 + jj) * KTOT;

  const float bi_i = biasc[0 * HSZ + hid0 + jj];
  const float bi_f = biasc[1 * HSZ + hid0 + jj];
  const float bi_g = biasc[2 * HSZ + hid0 + jj];
  const float bi_o = biasc[3 * HSZ + hid0 + jj];

  const int klo2     = (lane < 16) ? 0 : 16;   // A: byte offset of first K run
  const int kwb      = (lane < 16) ? 0 : 8;    // B: K offset of lane half
  const int mrow_off = (lane >> 4) << 3;       // C: M = vgpr + 8*(lane>=16)

  unsigned epoch = 0;
  grid_barrier(bar, epoch); ++epoch;           // weights/h0 visible everywhere

  const v8f vzero = {0.f, 0.f, 0.f, 0.f, 0.f, 0.f, 0.f, 0.f};

  for (int t = 0; t < T_STEPS; ++t) {
    const unsigned short* hprev = (t & 1) ? hbuf1 : hbuf0;
    unsigned short*       hnext = (t & 1) ? hbuf0 : hbuf1;
    const float* xt = inputs + (size_t)t * NBATCH * ISZ;

    if (t + 1 < T_STEPS)
      __builtin_prefetch(inputs + (size_t)(t + 1) * NBATCH * ISZ +
                         (size_t)tid * ISZ, 0, 1);

    v8f acc[2][4];
#pragma unroll
    for (int mt = 0; mt < 2; ++mt)
#pragma unroll
      for (int nt = 0; nt < 4; ++nt) acc[mt][nt] = vzero;

    // pre-stage chunk 0 (x, fp32 -> bf16) into buffer 0
    {
#pragma unroll
      for (int it = 0; it < 8; ++it) {
        int idx = tid + it * NTHREADS;
        int row = idx >> 3;
        int k4  = (idx & 7) << 2;
        float4 f = *(const float4*)(xt + (size_t)row * ISZ + k4);
        uint2 p; p.x = pack_bf16(f.x, f.y); p.y = pack_bf16(f.z, f.w);
        *(uint2*)(&smA[0][0] + row * A_STRIDE_B + k4 * 2) = p;
      }
    }
    __syncthreads();

    int pv = 0;
    for (int kb = 0; kb < KTOT; kb += KC) {
      const char* cur = &smA[pv][0];
      const int   nxt = pv ^ 1;
      const int   kn  = kb + KC;

      // stage next chunk into the other buffer (overlaps with WMMA below)
      if (kn < KTOT) {
        if (kn < ISZ) {                       // x chunk: fp32 -> bf16 via VALU
#pragma unroll
          for (int it = 0; it < 8; ++it) {
            int idx = tid + it * NTHREADS;
            int row = idx >> 3;
            int k4  = (idx & 7) << 2;
            float4 f = *(const float4*)(xt + (size_t)row * ISZ + kn + k4);
            uint2 p; p.x = pack_bf16(f.x, f.y); p.y = pack_bf16(f.z, f.w);
            *(uint2*)(&smA[nxt][0] + row * A_STRIDE_B + k4 * 2) = p;
          }
        } else {                              // h chunk: async bf16 copy to LDS
          const unsigned short* gsrc = hprev + (size_t)tid * HSZ + (kn - ISZ);
          unsigned ldst = smAbase[nxt] + (unsigned)tid * A_STRIDE_B;
#pragma unroll
          for (int s2 = 0; s2 < 4; ++s2)
            ASYNC_COPY_B128(ldst + s2 * 16, gsrc + s2 * 8);
        }
      }

      // A fragments (LDS): lane<16 K{0..7,16..23}, lane>=16 K{8..15,24..31}
      v16bf afr[2];
#pragma unroll
      for (int mt = 0; mt < 2; ++mt) {
        const char* ap = cur + (rb + mt * 16 + jj) * A_STRIDE_B + klo2;
        afr[mt] = join16(*(const v8bf*)ap, *(const v8bf*)(ap + 32));
      }
      // B fragments (bf16 weights, L2-resident): lane<16 K{0..7,16..23} else K{8..15,24..31}
      v16bf bfr[4];
#pragma unroll
      for (int nt = 0; nt < 4; ++nt) {
        const unsigned short* bp = wrow[nt] + kb + kwb;
        bfr[nt] = join16(*(const v8bf*)bp, *(const v8bf*)(bp + 16));
      }
#pragma unroll
      for (int mt = 0; mt < 2; ++mt)
#pragma unroll
        for (int nt = 0; nt < 4; ++nt)
          acc[mt][nt] = __builtin_amdgcn_wmma_f32_16x16x32_bf16(
              false, afr[mt], false, bfr[nt], (short)0, acc[mt][nt],
              false, false);

      async_wait0();                           // staged h chunk landed in LDS
      __syncthreads();                         // one barrier per chunk
      pv = nxt;
    }

    // fused LSTM cell update: nt 0..3 == i,f,g,o for hidden unit hid0+jj
#pragma unroll
    for (int mt = 0; mt < 2; ++mt) {
#pragma unroll
      for (int v = 0; v < 8; ++v) {
        int brow = rb + mt * 16 + mrow_off + v;
        float iv = sigm(acc[mt][0][v] + bi_i);
        float fv = sigm(acc[mt][1][v] + bi_f);
        float gv = tanh_fast(acc[mt][2][v] + bi_g);
        float ov = sigm(acc[mt][3][v] + bi_o);
        float c  = smC[brow * NH + jj];
        float cn = fv * c + iv * gv;
        smC[brow * NH + jj] = cn;
        hnext[(size_t)brow * HSZ + hid0 + jj] =
            (unsigned short)f2bf(ov * tanh_fast(cn));
      }
    }

    grid_barrier(bar, epoch); ++epoch;         // publish h for next step
  }
}

// ---------------------------------------------------------------------------
// Phase 2: out[b] = h_last[b,:] . W_dec + b_dec  (h_last is bf16 in hbuf0)
// ---------------------------------------------------------------------------
__global__ void lstm_decode(const unsigned short* __restrict__ h,
                            const float* __restrict__ Wd,
                            const float* __restrict__ bd,
                            float* __restrict__ out)
{
  int b = threadIdx.x;
  const unsigned short* hr = h + (size_t)b * HSZ;
  float s = 0.f;
  for (int j = 0; j < HSZ; ++j)
    s += __uint_as_float((unsigned)hr[j] << 16) * Wd[j];
  out[b] = s + bd[0];
}

extern "C" void kernel_launch(void* const* d_in, const int* in_sizes, int n_in,
                              void* d_out, int out_size, void* d_ws, size_t ws_size,
                              hipStream_t stream)
{
  const float* inputs = (const float*)d_in[0];
  const float* h0     = (const float*)d_in[1];
  const float* c0     = (const float*)d_in[2];
  const float* W_ih   = (const float*)d_in[3];
  const float* W_hh   = (const float*)d_in[4];
  const float* b_ih   = (const float*)d_in[5];
  const float* b_hh   = (const float*)d_in[6];
  const float* W_dec  = (const float*)d_in[7];
  const float* b_dec  = (const float*)d_in[8];
  float* out = (float*)d_out;

  // workspace (~3.5 MB): bf16 h ping-pong, bf16 weights, bias, barrier
  char* ws = (char*)d_ws;
  unsigned short* hbuf0 = (unsigned short*)ws;                           // 256 KB
  unsigned short* hbuf1 = (unsigned short*)(ws + (size_t)NBATCH * HSZ * 2);
  unsigned short* Wb    = (unsigned short*)(ws + (size_t)2 * NBATCH * HSZ * 2); // 3 MB
  float*          biasc = (float*)((char*)Wb + (size_t)GSZ * KTOT * 2);  // 8 KB
  unsigned*       bar   = (unsigned*)((char*)biasc + (size_t)GSZ * 4);

  lstm_prep<<<512, 256, 0, stream>>>(W_ih, W_hh, b_ih, b_hh, Wb, biasc, bar);
  lstm_persistent<<<NWG, NTHREADS, 0, stream>>>(inputs, h0, c0, Wb, biasc,
                                                hbuf0, hbuf1, bar);
  lstm_decode<<<1, NBATCH, 0, stream>>>(hbuf0, W_dec, b_dec, out);
}
